// RECT_L_82446192214207
// MI455X (gfx1250) — compile-verified, matrix operand order checked by reading, not run on hardware
//
#include <hip/hip_runtime.h>
#include <hip/hip_bf16.h>

// ---------------------------------------------------------------------------
// GCN layer on MI455X / gfx1250 (wave32):
//   preds = relu( Ahat @ (x*ns) @ Wg * nd + bg ) @ Wf^T + bf
// Strategy: bf16 WMMA (v_wmma_f32_16x16x32_bf16) for both GEMMs with all
// f32->bf16 conversion hoisted into bandwidth-bound elementwise passes;
// edge aggregation via hardware f32 atomics (L2-resident buffers).
// ---------------------------------------------------------------------------

#define NNODES   50000
#define NEDGES   1600000
#define INFEATS  512
#define NHID     128

typedef __attribute__((ext_vector_type(16))) __bf16 v16bf;
typedef __attribute__((ext_vector_type(8)))  float  v8f;

union FragAB {
    v16bf v;
    unsigned u[8];
    uint4 q[2];
};

// round-to-nearest-even pack of two f32 -> packed bf16x2 dword
__device__ __forceinline__ unsigned pack2_bf16(float a, float b) {
    unsigned ua = __float_as_uint(a);
    unsigned ub = __float_as_uint(b);
    ua += 0x7FFFu + ((ua >> 16) & 1u);
    ub += 0x7FFFu + ((ub >> 16) & 1u);
    return (ua >> 16) | (ub & 0xFFFF0000u);
}

// K-offset (in halves) inside a 16x32 bf16 fragment for packed pair index v
__device__ __forceinline__ int frag_koff(int v) { return (v < 4) ? (2 * v) : (8 + 2 * v); }

// ---------------------------------------------------------------------------
__global__ void zero_f32(float* p, int n) {
    int i = blockIdx.x * blockDim.x + threadIdx.x;
    if (i < n) p[i] = 0.0f;
}

__global__ void degree_kernel(const int* __restrict__ src, const int* __restrict__ dst,
                              float* deg_out, float* deg_in) {
    int e = blockIdx.x * blockDim.x + threadIdx.x;
    if (e < NEDGES) {
        unsafeAtomicAdd(&deg_out[src[e]], 1.0f);
        unsafeAtomicAdd(&deg_in[dst[e]], 1.0f);
    }
}

__global__ void norm_kernel(float* p, int n) {
    int i = blockIdx.x * blockDim.x + threadIdx.x;
    if (i < n) {
        float v = p[i];
        p[i] = (v > 0.0f) ? rsqrtf(v) : 0.0f;
    }
}

// xbf[n,k] = bf16( x[n,k] * norm_src[n] ), row-major, 8 elems/thread.
__global__ void scale_cvt_x(const float* __restrict__ X, const float* __restrict__ norm_src,
                            unsigned* __restrict__ XBF) {
    int i = blockIdx.x * blockDim.x + threadIdx.x;      // N*512/8 threads
    if (i >= NNODES * (INFEATS / 8)) return;
    int row = i >> 6;                                   // 512/8 = 64 chunks per row
    const float ns = norm_src[row];
    const float4* xp = (const float4*)(X + (size_t)i * 8);
    float4 a = xp[0], b = xp[1];
    uint4 o;
    o.x = pack2_bf16(a.x * ns, a.y * ns);
    o.y = pack2_bf16(a.z * ns, a.w * ns);
    o.z = pack2_bf16(b.x * ns, b.y * ns);
    o.w = pack2_bf16(b.z * ns, b.w * ns);
    ((uint4*)XBF)[i] = o;
}

// Pack gcn_w [512 x 128] row-major into per-lane B fragments (B[k][n] = W[k*128+n]).
__global__ void pack_b1(const float* __restrict__ W, unsigned* __restrict__ out) {
    int t = blockIdx.x * blockDim.x + threadIdx.x;       // 8 tiles * 16 ksteps * 32 lanes
    if (t >= 8 * 16 * 32) return;
    int lane = t & 31, ks = (t >> 5) & 15, tile = t >> 9;
    int n = tile * 16 + (lane & 15);
    int kb = ks * 32 + ((lane >> 4) << 3);
    unsigned* d = out + (size_t)t * 8;
#pragma unroll
    for (int v = 0; v < 8; ++v) {
        int k = kb + frag_koff(v);
        d[v] = pack2_bf16(W[(size_t)k * NHID + n], W[(size_t)(k + 1) * NHID + n]);
    }
}

// Pack fc_w [512 x 128] (out,in) so B[k][n] = fc_w[n*128 + k]  (fc_w^T).
__global__ void pack_b2(const float* __restrict__ W, unsigned* __restrict__ out) {
    int t = blockIdx.x * blockDim.x + threadIdx.x;       // 32 tiles * 4 ksteps * 32 lanes
    if (t >= 32 * 4 * 32) return;
    int lane = t & 31, ks = (t >> 5) & 3, tile = t >> 7;
    int n = tile * 16 + (lane & 15);
    int kb = ks * 32 + ((lane >> 4) << 3);
    const float* wrow = W + (size_t)n * NHID;
    unsigned* d = out + (size_t)t * 8;
#pragma unroll
    for (int v = 0; v < 8; ++v) {
        int k = kb + frag_koff(v);
        d[v] = pack2_bf16(wrow[k], wrow[k + 1]);
    }
}

// ---------------------------------------------------------------------------
// GEMM1: xw[50000,128] = xbf @ gcn_w.  Block = 256 (8 waves); wave w owns the
// 16x16 tile at cols w*16; blockIdx.x = M-tile (50000/16 = 3125 exact).
// Row-major bf16 A fragment = two global_load_b128 per lane (ISA 7.12.2).
__global__ void __launch_bounds__(256) gemm1_xw(const unsigned* __restrict__ XBF,
                                                const unsigned* __restrict__ B1,
                                                float* __restrict__ XW) {
    const int lane = threadIdx.x & 31;
    const int wave = threadIdx.x >> 5;
    const int l15  = lane & 15;
    const int half = lane >> 4;
    const int m0   = blockIdx.x * 16;
    const int n0   = wave * 16;

    // per-lane A row base, in halves: row m0+l15, K offset half*8
    const unsigned* arow = XBF + ((size_t)(m0 + l15) * INFEATS + (half << 3)) / 2;

    v8f acc = {};
#pragma unroll 4
    for (int ks = 0; ks < 16; ++ks) {
        FragAB a;
        const uint4* ap = (const uint4*)(arow + ks * 16);   // ks*32 halves
        a.q[0] = ap[0];                                     // K +0..7
        a.q[1] = ap[2];                                     // K +16..23 (skip other half-wave's 8)
        FragAB b;
        const uint4* bp = (const uint4*)(B1 + ((size_t)(wave * 16 + ks) * 32 + lane) * 8);
        b.q[0] = bp[0];
        b.q[1] = bp[1];
        acc = __builtin_amdgcn_wmma_f32_16x16x32_bf16(false, a.v, false, b.v,
                                                      (short)0, acc, false, false);
    }

    const int mrow = m0 + (half << 3);
    float* outp = XW + (size_t)mrow * NHID + n0 + l15;
#pragma unroll
    for (int r = 0; r < 8; ++r) outp[(size_t)r * NHID] = acc[r];
}

// ---------------------------------------------------------------------------
// Edge aggregation: agg[dst] += xw[src].  One thread per (edge, 4 features).
// xw/agg (25.6 MB each) are L2-resident -> gather + HW f32 atomics stay on-chip.
__global__ void scatter_agg(const float* __restrict__ xw, const int* __restrict__ src,
                            const int* __restrict__ dst, float* __restrict__ agg) {
    long long gid = (long long)blockIdx.x * blockDim.x + threadIdx.x;
    int e = (int)(gid >> 5);
    if (e >= NEDGES) return;
    int q = ((int)gid & 31) << 2;
    int s = src[e], d = dst[e];
    const float4 v = *(const float4*)(xw + (size_t)s * NHID + q);
    float* ap = agg + (size_t)d * NHID + q;
    unsafeAtomicAdd(ap + 0, v.x);
    unsafeAtomicAdd(ap + 1, v.y);
    unsafeAtomicAdd(ap + 2, v.z);
    unsafeAtomicAdd(ap + 3, v.w);
}

// h1bf = bf16( relu(agg * norm_dst + gcn_b) ), row-major, 8 elems/thread.
__global__ void relu_bias_cvt(const float* __restrict__ agg, const float* __restrict__ norm_dst,
                              const float* __restrict__ gcn_b, unsigned* __restrict__ H1BF) {
    int i = blockIdx.x * blockDim.x + threadIdx.x;      // N*128/8 threads
    if (i >= NNODES * (NHID / 8)) return;
    int row = i >> 4;                                   // 128/8 = 16 chunks per row
    int col = (i & 15) << 3;
    const float nd = norm_dst[row];
    const float4* ap = (const float4*)(agg + (size_t)i * 8);
    const float4* bp = (const float4*)(gcn_b + col);
    float4 a = ap[0], b = ap[1];
    float4 c0 = bp[0], c1 = bp[1];
    float r0 = fmaxf(a.x * nd + c0.x, 0.0f), r1 = fmaxf(a.y * nd + c0.y, 0.0f);
    float r2 = fmaxf(a.z * nd + c0.z, 0.0f), r3 = fmaxf(a.w * nd + c0.w, 0.0f);
    float r4 = fmaxf(b.x * nd + c1.x, 0.0f), r5 = fmaxf(b.y * nd + c1.y, 0.0f);
    float r6 = fmaxf(b.z * nd + c1.z, 0.0f), r7 = fmaxf(b.w * nd + c1.w, 0.0f);
    uint4 o;
    o.x = pack2_bf16(r0, r1);
    o.y = pack2_bf16(r2, r3);
    o.z = pack2_bf16(r4, r5);
    o.w = pack2_bf16(r6, r7);
    ((uint4*)H1BF)[i] = o;
}

// ---------------------------------------------------------------------------
// GEMM2: preds[50000,512] = h1 @ fc_w^T + fc_b.  grid = (3125, 4); wave w
// covers cols blockIdx.y*128 + w*16; K = 128 -> 4 WMMA steps.
__global__ void __launch_bounds__(256) gemm2_fc(const unsigned* __restrict__ H1BF,
                                                const unsigned* __restrict__ B2,
                                                const float* __restrict__ fc_b,
                                                float* __restrict__ OUT) {
    const int lane = threadIdx.x & 31;
    const int wave = threadIdx.x >> 5;
    const int l15  = lane & 15;
    const int half = lane >> 4;
    const int m0   = blockIdx.x * 16;
    const int tileN = blockIdx.y * 8 + wave;   // 0..31
    const int n0   = tileN * 16;

    const unsigned* arow = H1BF + ((size_t)(m0 + l15) * NHID + (half << 3)) / 2;

    v8f acc = {};
#pragma unroll
    for (int ks = 0; ks < 4; ++ks) {
        FragAB a;
        const uint4* ap = (const uint4*)(arow + ks * 16);
        a.q[0] = ap[0];
        a.q[1] = ap[2];
        FragAB b;
        const uint4* bp = (const uint4*)(B2 + ((size_t)(tileN * 4 + ks) * 32 + lane) * 8);
        b.q[0] = bp[0];
        b.q[1] = bp[1];
        acc = __builtin_amdgcn_wmma_f32_16x16x32_bf16(false, a.v, false, b.v,
                                                      (short)0, acc, false, false);
    }

    const float bias = fc_b[n0 + l15];
    const int mrow = m0 + (half << 3);
    float* outp = OUT + (size_t)mrow * INFEATS + n0 + l15;
#pragma unroll
    for (int r = 0; r < 8; ++r) outp[(size_t)r * INFEATS] = acc[r] + bias;
}

// ---------------------------------------------------------------------------
extern "C" void kernel_launch(void* const* d_in, const int* in_sizes, int n_in,
                              void* d_out, int out_size, void* d_ws, size_t ws_size,
                              hipStream_t stream) {
    const float* x     = (const float*)d_in[0];
    const int*   src   = (const int*)d_in[1];
    const int*   dst   = (const int*)d_in[2];
    const float* gcn_w = (const float*)d_in[3];
    const float* gcn_b = (const float*)d_in[4];
    const float* fc_w  = (const float*)d_in[5];
    const float* fc_b  = (const float*)d_in[6];
    float* out = (float*)d_out;

    // Workspace layout. Region R (51.2 MB) first holds xbf (bf16 x*ns); after
    // GEMM1 consumes it, the same region is reused for agg (f32) + h1bf (bf16).
    float*    norm_src = (float*)d_ws;                              // N
    float*    norm_dst = norm_src + NNODES;                         // N
    float*    xw       = norm_dst + NNODES;                         // N*128 f32
    float*    regionR  = xw + (size_t)NNODES * NHID;                // 51.2 MB
    unsigned* xbf      = (unsigned*)regionR;                        // N*512 bf16
    float*    agg      = regionR;                                   // N*128 f32 (reuse)
    unsigned* h1bf     = (unsigned*)(agg + (size_t)NNODES * NHID);  // N*128 bf16 (reuse)
    unsigned* b1       = (unsigned*)((char*)regionR + (size_t)NNODES * INFEATS * 2);
    unsigned* b2       = b1 + 8 * 16 * 32 * 8;

    // 1) degrees -> norms
    zero_f32<<<(2 * NNODES + 255) / 256, 256, 0, stream>>>(norm_src, 2 * NNODES);
    degree_kernel<<<NEDGES / 256, 256, 0, stream>>>(src, dst, norm_src, norm_dst);
    norm_kernel<<<(2 * NNODES + 255) / 256, 256, 0, stream>>>(norm_src, 2 * NNODES);

    // 2) prepack weights into WMMA B-fragment layout; pre-scale+convert x
    pack_b1<<<16, 256, 0, stream>>>(gcn_w, b1);
    pack_b2<<<16, 256, 0, stream>>>(fc_w, b2);
    scale_cvt_x<<<NNODES * (INFEATS / 8) / 256, 256, 0, stream>>>(x, norm_src, xbf);

    // 3) xw = xbf @ gcn_w
    gemm1_xw<<<NNODES / 16, 256, 0, stream>>>(xbf, b1, xw);

    // 4) agg = scatter_add(xw[src] -> dst)   (agg aliases xbf region; xbf dead)
    zero_f32<<<(NNODES * NHID) / 256, 256, 0, stream>>>(agg, NNODES * NHID);
    scatter_agg<<<(NEDGES * 32) / 256, 256, 0, stream>>>(xw, src, dst, agg);

    // 5) h1bf = bf16(relu(agg * norm_dst + gcn_b))
    relu_bias_cvt<<<NNODES * (NHID / 8) / 256, 256, 0, stream>>>(agg, norm_dst, gcn_b, h1bf);

    // 6) preds = h1 @ fc_w^T + fc_b
    gemm2_fc<<<dim3(NNODES / 16, 4), 256, 0, stream>>>(h1bf, b2, fc_b, out);
}
